// CustomTiledAttentionNoOnlineSoftmaxPyCore_49452253446943
// MI455X (gfx1250) — matrix operand order repaired and checked by reading, not compiled, not run on hardware
//
#include <hip/hip_runtime.h>

typedef _Float16 h16v __attribute__((ext_vector_type(16)));
typedef _Float16 h8v  __attribute__((ext_vector_type(8)));
typedef float    f8v  __attribute__((ext_vector_type(8)));
typedef float    f4v  __attribute__((ext_vector_type(4)));

#define S_LEN 2048
#define D_DIM 64
#define QTILE 128
#define KTILE 32

static __device__ __forceinline__ h16v cat8(h8v a, h8v b) {
  return __builtin_shufflevector(a, b, 0, 1, 2, 3, 4, 5, 6, 7,
                                 8, 9, 10, 11, 12, 13, 14, 15);
}

static __device__ __forceinline__ h16v lds_frag(const _Float16* base) {
  return cat8(*(const h8v*)base, *(const h8v*)(base + 16));
}

// Low 32 bits of a flat shared-memory address are the LDS byte offset
// (the shared aperture lives in addr[63:32]).
static __device__ __forceinline__ unsigned lds_off(const void* p) {
  return (unsigned)(uintptr_t)p;
}

// Async DMA: global (SGPR base + per-lane 32-bit offset) -> LDS, 16B per lane.
// Tracked by ASYNCcnt; no VGPR data, no VALU.
static __device__ __forceinline__ void async_ld_b128(unsigned lds,
                                                     unsigned voff,
                                                     const void* base) {
  asm volatile("global_load_async_to_lds_b128 %0, %1, %2"
               :: "v"(lds), "v"(voff), "s"(base)
               : "memory");
}

static __device__ __forceinline__ void wait_async0() {
  asm volatile("s_wait_asynccnt 0x0" ::: "memory");
}

// ---------------------------------------------------------------------------
// Pre-pass 1: K fp32 -> fp16, same [bh][s][d] layout.
__global__ __launch_bounds__(256) void cvt_f16_kernel(
    const float* __restrict__ src, _Float16* __restrict__ dst) {
  const size_t idx = ((size_t)blockIdx.x * 256 + threadIdx.x) * 8;
  const float* s = src + idx;
  f4v a = *(const f4v*)s;
  f4v b = *(const f4v*)(s + 4);
  h8v o;
#pragma unroll
  for (int j = 0; j < 4; ++j) {
    o[j]     = (_Float16)a[j];
    o[4 + j] = (_Float16)b[j];
  }
  *(h8v*)(dst + idx) = o;
}

// Pre-pass 2: V fp32 [bh][s][d] -> fp16 transposed [bh][d][s], 64x64 LDS tiles.
__global__ __launch_bounds__(256) void vtrans_f16_kernel(
    const float* __restrict__ V, _Float16* __restrict__ Vth) {
  __shared__ alignas(16) _Float16 T[64][72];  // 72: pad to dodge bank conflicts
  const int nst = S_LEN / 64;
  const int bh = blockIdx.x / nst;
  const int st = blockIdx.x % nst;
  const int row = threadIdx.x >> 2;        // 0..63
  const int cg  = (threadIdx.x & 3) * 16;  // 16-elem column group

  const float* p = V + ((size_t)bh * S_LEN + st * 64 + row) * D_DIM + cg;
#pragma unroll
  for (int j = 0; j < 4; ++j) {
    f4v x = *(const f4v*)(p + j * 4);
#pragma unroll
    for (int e = 0; e < 4; ++e) T[cg + j * 4 + e][row] = (_Float16)x[e];
  }
  __syncthreads();
  // row now indexes d; cg indexes s within the tile. Coalesced 32B stores.
  _Float16* dst = Vth + ((size_t)bh * D_DIM + row) * S_LEN + st * 64 + cg;
  *(h8v*)dst       = *(const h8v*)&T[row][cg];
  *(h8v*)(dst + 8) = *(const h8v*)&T[row][cg + 8];
}

// ---------------------------------------------------------------------------
// Main flash-attention kernel, async-DMA LDS staging (f16 K / V^T in d_ws).
__global__ __launch_bounds__(256) void fa_fwd_async_kernel(
    const float* __restrict__ Q, const _Float16* __restrict__ Kh,
    const _Float16* __restrict__ Vth, float* __restrict__ O) {
  __shared__ alignas(16) _Float16 Ks[2][KTILE * D_DIM];  // [k][d]
  __shared__ alignas(16) _Float16 Vt[2][D_DIM * KTILE];  // [d][k]

  const int tid  = threadIdx.x;
  const int lane = tid & 31;
  const int wid  = tid >> 5;
  const bool hi  = (lane >= 16);
  const int l15  = lane & 15;
  const int b    = hi ? 8 : 0;

  const int ntq = S_LEN / QTILE;
  const int qt  = blockIdx.x % ntq;
  const int bh  = blockIdx.x / ntq;

  const float*    qb  = Q   + (size_t)bh * S_LEN * D_DIM;
  const _Float16* khb = Kh  + (size_t)bh * S_LEN * D_DIM;
  const _Float16* vtb = Vth + (size_t)bh * D_DIM * S_LEN;
  float*          ob  = O   + (size_t)bh * S_LEN * D_DIM;

  const int qrow = qt * QTILE + wid * 16 + l15;

  // Q fragment (B operand), softmax scale folded in (log2 domain).
  const float qscl = 0.125f * 1.4426950408889634f;
  h16v bq[2];
#pragma unroll
  for (int g = 0; g < 2; ++g) {
    const float* src = qb + (size_t)qrow * D_DIM + g * 32 + (hi ? 8 : 0);
#pragma unroll
    for (int e = 0; e < 16; ++e) bq[g][e] = (_Float16)(src[e + (e & 8)] * qscl);
  }

  // Row remap: tile0 k={0..7,16..23}, tile1 k={8..15,24..31} so S^T C-layout
  // coincides with the P^T B-layout (no cross-lane swaps).
  const int row0 = l15 + (l15 & 8);
  const int row1 = row0 + 8;

  f8v ot[4] = {};
  float m_run = -INFINITY;
  float l_run = 0.0f;

  // Async staging assignments: one B128 per thread per tile per matrix.
  const int kr = tid >> 3;             // K tile row 0..31
  const int kc = (tid & 7) * 8;        // K tile 8-half chunk
  const int vr = tid >> 2;             // V^T tile row (d) 0..63
  const int vc = (tid & 3) * 8;        // V^T tile 8-half chunk

  const unsigned ldsK[2] = {lds_off(&Ks[0][kr * D_DIM + kc]),
                            lds_off(&Ks[1][kr * D_DIM + kc])};
  const unsigned ldsV[2] = {lds_off(&Vt[0][vr * KTILE + vc]),
                            lds_off(&Vt[1][vr * KTILE + vc])};
  const unsigned voffK = (unsigned)((kr * D_DIM + kc) * 2);   // + k0*D*2
  const unsigned voffV = (unsigned)((vr * S_LEN + vc) * 2);   // + k0*2

  auto issue_tile = [&](int k0, int p) {
    async_ld_b128(ldsK[p], voffK + (unsigned)(k0 * D_DIM * 2), khb);
    async_ld_b128(ldsV[p], voffV + (unsigned)(k0 * 2), vtb);
  };

  auto compute_tile = [&](const _Float16* KsB, const _Float16* VtB) {
    h16v a00 = lds_frag(&KsB[row0 * D_DIM + b]);
    h16v a10 = lds_frag(&KsB[row1 * D_DIM + b]);
    h16v a01 = lds_frag(&KsB[row0 * D_DIM + 32 + b]);
    h16v a11 = lds_frag(&KsB[row1 * D_DIM + 32 + b]);
    f8v st0 = {};
    f8v st1 = {};
    st0 = __builtin_amdgcn_wmma_f32_16x16x32_f16(false, a00, false, bq[0],
                                                 (short)0, st0, false, false);
    st1 = __builtin_amdgcn_wmma_f32_16x16x32_f16(false, a10, false, bq[0],
                                                 (short)0, st1, false, false);
    st0 = __builtin_amdgcn_wmma_f32_16x16x32_f16(false, a01, false, bq[1],
                                                 (short)0, st0, false, false);
    st1 = __builtin_amdgcn_wmma_f32_16x16x32_f16(false, a11, false, bq[1],
                                                 (short)0, st1, false, false);

    h16v av[4];
#pragma unroll
    for (int j = 0; j < 4; ++j) {
      const int d = j * 16 + l15;
      av[j] = cat8(*(const h8v*)&VtB[d * KTILE + 2 * b], *(const h8v*)&VtB[d * KTILE + 2 * b + 8]);
    }

    float mt = -INFINITY;
#pragma unroll
    for (int i = 0; i < 8; ++i) mt = fmaxf(mt, fmaxf(st0[i], st1[i]));
    mt = fmaxf(mt, __shfl_xor(mt, 16, 32));
    const float m_new = fmaxf(m_run, mt);
    const float alpha = __builtin_amdgcn_exp2f(m_run - m_new);

    float p0[8], p1[8];
    float rs = 0.0f;
#pragma unroll
    for (int i = 0; i < 8; ++i) {
      p0[i] = __builtin_amdgcn_exp2f(st0[i] - m_new);
      p1[i] = __builtin_amdgcn_exp2f(st1[i] - m_new);
      rs += p0[i] + p1[i];
    }
    rs += __shfl_xor(rs, 16, 32);
    l_run = l_run * alpha + rs;
    m_run = m_new;
#pragma unroll
    for (int j = 0; j < 4; ++j)
#pragma unroll
      for (int i = 0; i < 8; ++i) ot[j][i] *= alpha;

    h16v pt;
#pragma unroll
    for (int i = 0; i < 8; ++i) {
      pt[i]     = (_Float16)p0[i];
      pt[8 + i] = (_Float16)p1[i];
    }
#pragma unroll
    for (int j = 0; j < 4; ++j) {
      ot[j] = __builtin_amdgcn_wmma_f32_16x16x32_f16(false, av[j], false, pt,
                                                     (short)0, ot[j], false,
                                                     false);
    }
  };

  issue_tile(0, 0);
#pragma unroll 1
  for (int k0 = 0; k0 < S_LEN; k0 += 2 * KTILE) {
    wait_async0();
    __syncthreads();
    if (k0 + KTILE < S_LEN) issue_tile(k0 + KTILE, 1);
    compute_tile(Ks[0], Vt[0]);

    wait_async0();
    __syncthreads();
    if (k0 + 2 * KTILE < S_LEN) issue_tile(k0 + 2 * KTILE, 0);
    compute_tile(Ks[1], Vt[1]);
  }

  const float inv_l = 1.0f / l_run;
#pragma unroll
  for (int j = 0; j < 4; ++j) {
#pragma unroll
    for (int i = 0; i < 8; ++i) {
      const int d = j * 16 + i + (hi ? 8 : 0);
      ob[(size_t)qrow * D_DIM + d] = ot[j][i] * inv_l;
    }
  }
}

// ---------------------------------------------------------------------------
// Fallback (no workspace): register-staged fp32->fp16 conversion in-kernel.
__global__ __launch_bounds__(256) void fa_fwd_fallback_kernel(
    const float* __restrict__ Q, const float* __restrict__ K,
    const float* __restrict__ V, float* __restrict__ O) {
  __shared__ alignas(16) _Float16 Ks[KTILE * D_DIM];
  __shared__ alignas(16) _Float16 Vt[D_DIM * KTILE];

  const int tid  = threadIdx.x;
  const int lane = tid & 31;
  const int wid  = tid >> 5;
  const bool hi  = (lane >= 16);
  const int l15  = lane & 15;
  const int b    = hi ? 8 : 0;

  const int ntq = S_LEN / QTILE;
  const int qt  = blockIdx.x % ntq;
  const int bh  = blockIdx.x / ntq;

  const float* qb = Q + (size_t)bh * S_LEN * D_DIM;
  const float* kb = K + (size_t)bh * S_LEN * D_DIM;
  const float* vb = V + (size_t)bh * S_LEN * D_DIM;
  float*       ob = O + (size_t)bh * S_LEN * D_DIM;

  const int qrow = qt * QTILE + wid * 16 + l15;

  const float qscl = 0.125f * 1.4426950408889634f;
  h16v bq[2];
#pragma unroll
  for (int g = 0; g < 2; ++g) {
    const float* src = qb + (size_t)qrow * D_DIM + g * 32 + (hi ? 8 : 0);
#pragma unroll
    for (int e = 0; e < 16; ++e) bq[g][e] = (_Float16)(src[e + (e & 8)] * qscl);
  }

  const int row0 = l15 + (l15 & 8);
  const int row1 = row0 + 8;

  f8v ot[4] = {};
  float m_run = -INFINITY;
  float l_run = 0.0f;

  const int r  = tid >> 3;
  const int c8 = (tid & 7) * 8;

  const float* kst0 = kb + (size_t)r * D_DIM + c8;
  const float* vst0 = vb + (size_t)r * D_DIM + c8;
  f4v ka = *(const f4v*)kst0;
  f4v kc = *(const f4v*)(kst0 + 4);
  f4v va = *(const f4v*)vst0;
  f4v vc = *(const f4v*)(vst0 + 4);

  for (int k0 = 0; k0 < S_LEN; k0 += KTILE) {
    __syncthreads();
    h8v kp;
#pragma unroll
    for (int j = 0; j < 4; ++j) {
      kp[j]     = (_Float16)ka[j];
      kp[4 + j] = (_Float16)kc[j];
    }
    *(h8v*)&Ks[r * D_DIM + c8] = kp;
#pragma unroll
    for (int j = 0; j < 4; ++j) {
      Vt[(c8 + j) * KTILE + r]     = (_Float16)va[j];
      Vt[(c8 + 4 + j) * KTILE + r] = (_Float16)vc[j];
    }
    __syncthreads();

    if (k0 + KTILE < S_LEN) {
      const float* kn = kb + (size_t)(k0 + KTILE + r) * D_DIM + c8;
      const float* vn = vb + (size_t)(k0 + KTILE + r) * D_DIM + c8;
      ka = *(const f4v*)kn;
      kc = *(const f4v*)(kn + 4);
      va = *(const f4v*)vn;
      vc = *(const f4v*)(vn + 4);
      if (k0 + 2 * KTILE < S_LEN) {
        __builtin_prefetch(kn + (size_t)KTILE * D_DIM, 0, 1);
        __builtin_prefetch(vn + (size_t)KTILE * D_DIM, 0, 1);
      }
    }

    h16v a00 = lds_frag(&Ks[row0 * D_DIM + b]);
    h16v a10 = lds_frag(&Ks[row1 * D_DIM + b]);
    h16v a01 = lds_frag(&Ks[row0 * D_DIM + 32 + b]);
    h16v a11 = lds_frag(&Ks[row1 * D_DIM + 32 + b]);
    f8v st0 = {};
    f8v st1 = {};
    st0 = __builtin_amdgcn_wmma_f32_16x16x32_f16(false, a00, false, bq[0],
                                                 (short)0, st0, false, false);
    st1 = __builtin_amdgcn_wmma_f32_16x16x32_f16(false, a10, false, bq[0],
                                                 (short)0, st1, false, false);
    st0 = __builtin_amdgcn_wmma_f32_16x16x32_f16(false, a01, false, bq[1],
                                                 (short)0, st0, false, false);
    st1 = __builtin_amdgcn_wmma_f32_16x16x32_f16(false, a11, false, bq[1],
                                                 (short)0, st1, false, false);

    h16v av[4];
#pragma unroll
    for (int j = 0; j < 4; ++j) {
      const int d = j * 16 + l15;
      av[j] = cat8(*(const h8v*)&Vt[d * KTILE + 2 * b], *(const h8v*)&Vt[d * KTILE + 2 * b + 8]);
    }

    float mt = -INFINITY;
#pragma unroll
    for (int i = 0; i < 8; ++i) mt = fmaxf(mt, fmaxf(st0[i], st1[i]));
    mt = fmaxf(mt, __shfl_xor(mt, 16, 32));
    const float m_new = fmaxf(m_run, mt);
    const float alpha = __builtin_amdgcn_exp2f(m_run - m_new);

    float p0[8], p1[8];
    float rs = 0.0f;
#pragma unroll
    for (int i = 0; i < 8; ++i) {
      p0[i] = __builtin_amdgcn_exp2f(st0[i] - m_new);
      p1[i] = __builtin_amdgcn_exp2f(st1[i] - m_new);
      rs += p0[i] + p1[i];
    }
    rs += __shfl_xor(rs, 16, 32);
    l_run = l_run * alpha + rs;
    m_run = m_new;
#pragma unroll
    for (int j = 0; j < 4; ++j)
#pragma unroll
      for (int i = 0; i < 8; ++i) ot[j][i] *= alpha;

    h16v pt;
#pragma unroll
    for (int i = 0; i < 8; ++i) {
      pt[i]     = (_Float16)p0[i];
      pt[8 + i] = (_Float16)p1[i];
    }
#pragma unroll
    for (int j = 0; j < 4; ++j) {
      ot[j] = __builtin_amdgcn_wmma_f32_16x16x32_f16(false, av[j], false, pt,
                                                     (short)0, ot[j], false,
                                                     false);
    }
  }

  const float inv_l = 1.0f / l_run;
#pragma unroll
  for (int j = 0; j < 4; ++j) {
#pragma unroll
    for (int i = 0; i < 8; ++i) {
      const int d = j * 16 + i + (hi ? 8 : 0);
      ob[(size_t)qrow * D_DIM + d] = ot[j][i] * inv_l;
    }
  }
}

extern "C" void kernel_launch(void* const* d_in, const int* in_sizes, int n_in,
                              void* d_out, int out_size, void* d_ws,
                              size_t ws_size, hipStream_t stream) {
  const float* q = (const float*)d_in[0];
  const float* k = (const float*)d_in[1];
  const float* v = (const float*)d_in[2];
  float* o = (float*)d_out;

  const int bh_count = in_sizes[0] / (S_LEN * D_DIM);  // B*H = 32
  const int nblocks = bh_count * (S_LEN / QTILE);      // 512

  const size_t half_bytes =
      (size_t)bh_count * S_LEN * D_DIM * sizeof(_Float16);  // 8.4 MB each

  if (d_ws != nullptr && ws_size >= 2 * half_bytes) {
    _Float16* kh  = (_Float16*)d_ws;
    _Float16* vth = (_Float16*)((char*)d_ws + half_bytes);
    const int n8blocks = (int)((size_t)bh_count * S_LEN * D_DIM / 8 / 256);
    cvt_f16_kernel<<<n8blocks, 256, 0, stream>>>(k, kh);
    vtrans_f16_kernel<<<bh_count * (S_LEN / 64), 256, 0, stream>>>(v, vth);
    fa_fwd_async_kernel<<<nblocks, 256, 0, stream>>>(q, kh, vth, o);
  } else {
    fa_fwd_fallback_kernel<<<nblocks, 256, 0, stream>>>(q, k, v, o);
  }
}
